// Attention_77163382440391
// MI455X (gfx1250) — matrix operand order repaired and checked by reading, not compiled, not run on hardware
//
#include <hip/hip_runtime.h>

typedef __attribute__((ext_vector_type(16))) __bf16 v16bf;
typedef __attribute__((ext_vector_type(8)))  float  v8f;

#define BATCH   4
#define CCH     512
#define NTOK    1568          // 8*14*14
#define DHEAD   64
#define NHEADS  8
#define BR      128           // query rows per workgroup (8 waves x 16 rows)
#define BC      64            // key block
#define NKB     ((NTOK + BC - 1) / BC)   // 25
#define NRB     ((NTOK + BR - 1) / BR)   // 13
#define QSCALE  0.015625f     // s^2 = 1/64 folded entirely into Q (exact pow2)

#define QSTR 72               // LDS row strides (bf16 elems), even + padded vs 64
#define KSTR 72
#define VSTR 72
#define PSTR 72

__device__ __forceinline__ unsigned pk2(float a, float b) {
    unsigned short ua = __builtin_bit_cast(unsigned short, (__bf16)a);
    unsigned short ub = __builtin_bit_cast(unsigned short, (__bf16)b);
    return (unsigned)ua | ((unsigned)ub << 16);
}

// Stage one 64x64 K/V tile: kt row-major [key][d], vt transposed [d][key].
// Same (unscaled) values; layouts differ so both GEMMs read packed dwords.
__device__ __forceinline__ void stage_kv(__bf16* __restrict__ kt,
                                         __bf16* __restrict__ vt,
                                         const float* __restrict__ xb,
                                         int head, int key0, int tid) {
    if (key0 + BC <= NTOK) {            // full tile: float4 (b128) loads
#pragma unroll
        for (int i = 0; i < (BC * DHEAD) / (4 * 256); ++i) {   // 4 iters
            int idx = tid + i * 256;
            int d  = idx >> 4;          // 0..63
            int r4 = (idx & 15) * 4;    // 0..60
            float4 v = *(const float4*)&xb[(size_t)(d * NHEADS + head) * NTOK + key0 + r4];
            kt[(r4 + 0) * KSTR + d] = (__bf16)v.x;
            kt[(r4 + 1) * KSTR + d] = (__bf16)v.y;
            kt[(r4 + 2) * KSTR + d] = (__bf16)v.z;
            kt[(r4 + 3) * KSTR + d] = (__bf16)v.w;
            *(unsigned*)&vt[d * VSTR + r4]     = pk2(v.x, v.y);
            *(unsigned*)&vt[d * VSTR + r4 + 2] = pk2(v.z, v.w);
        }
    } else {                            // tail block: scalar, zero-filled
#pragma unroll
        for (int i = 0; i < (BC * DHEAD) / 256; ++i) {         // 16 iters
            int idx = tid + i * 256;
            int d = idx >> 6;
            int r = idx & 63;
            int n = key0 + r;
            float v = (n < NTOK) ? xb[(size_t)(d * NHEADS + head) * NTOK + n] : 0.0f;
            kt[r * KSTR + d] = (__bf16)v;
            vt[d * VSTR + r] = (__bf16)v;
        }
    }
}

__global__ __launch_bounds__(256)
void Attention_77163382440391_kernel(const float* __restrict__ x,
                                     float* __restrict__ out) {
    const int blk  = blockIdx.x;
    const int rb   = blk % NRB;
    const int bh   = blk / NRB;
    const int head = bh % NHEADS;
    const int b    = bh / NHEADS;

    const float* __restrict__ xb = x   + (size_t)b * CCH * NTOK;
    float*       __restrict__ ob = out + (size_t)b * CCH * NTOK;

    const int tid  = threadIdx.x;
    const int wave = tid >> 5;
    const int lane = tid & 31;
    const int lo16 = lane & 15;
    const int hi   = lane >> 4;

    // q[head][n][d] = x[b][(d*8+head)][n]   (n-contiguous columns)
    __shared__ __align__(16) __bf16 qt[BR * QSTR];         // Q * 1/64, [row][d]
    __shared__ __align__(16) __bf16 kt[2][BC * KSTR];      // K, [key][d], double-buffered
    __shared__ __align__(16) __bf16 vt[2][DHEAD * VSTR];   // V, [d][key], double-buffered
    __shared__ __align__(16) __bf16 pt[8 * 16 * PSTR];     // per-wave P patches

    const int row0 = rb * BR;

    // ---- stage Q tile (float4 fast path; scalar clamped tail path) ----
    if (row0 + BR <= NTOK) {
#pragma unroll
        for (int i = 0; i < (BR * DHEAD) / (4 * 256); ++i) {   // 8 iters
            int idx = tid + i * 256;
            int d  = idx >> 5;           // 0..63
            int r4 = (idx & 31) * 4;     // 0..124
            float4 v = *(const float4*)&xb[(size_t)(d * NHEADS + head) * NTOK + row0 + r4];
            qt[(r4 + 0) * QSTR + d] = (__bf16)(v.x * QSCALE);
            qt[(r4 + 1) * QSTR + d] = (__bf16)(v.y * QSCALE);
            qt[(r4 + 2) * QSTR + d] = (__bf16)(v.z * QSCALE);
            qt[(r4 + 3) * QSTR + d] = (__bf16)(v.w * QSCALE);
        }
    } else {
#pragma unroll
        for (int i = 0; i < (BR * DHEAD) / 256; ++i) {         // 32 iters
            int idx = tid + i * 256;
            int d = idx >> 7;
            int r = idx & 127;
            int n = row0 + r; if (n >= NTOK) n = NTOK - 1;     // clamp tail rows
            float v = xb[(size_t)(d * NHEADS + head) * NTOK + n];
            qt[r * QSTR + d] = (__bf16)(v * QSCALE);
        }
    }

    // prologue: stage first K/V block into buffer 0
    stage_kv(&kt[0][0], &vt[0][0], xb, head, 0, tid);
    __syncthreads();

    // ---- per-wave Q A-fragments (16 rows x 64 d = 2 chunks of K=32) ----
    v16bf aq[2];
    {
        union { v16bf v; unsigned u[8]; } tmp;
#pragma unroll
        for (int c = 0; c < 2; ++c) {
#pragma unroll
            for (int j = 0; j < 8; ++j) {
                int d = c * 32 + (j & 3) * 2 + (j >> 2) * 16 + hi * 8;
                tmp.u[j] = *(const unsigned*)&qt[(wave * 16 + lo16) * QSTR + d];
            }
            aq[c] = tmp.v;
        }
    }

    // online-softmax state: each lane tracks the 8 rows of its half-wave
    float mrow[8], lrow[8];
    v8f oacc[4];
#pragma unroll
    for (int r = 0; r < 8; ++r) { mrow[r] = -3.0e38f; lrow[r] = 0.0f; }
#pragma unroll
    for (int t = 0; t < 4; ++t)
#pragma unroll
        for (int r = 0; r < 8; ++r) oacc[t][r] = 0.0f;

    for (int kb = 0; kb < NKB; ++kb) {
        const int key0 = kb * BC;
        const int buf  = kb & 1;
        // software pipeline: kick off next block's staging before compute
        if (kb + 1 < NKB)
            stage_kv(&kt[buf ^ 1][0], &vt[buf ^ 1][0], xb, head, key0 + BC, tid);

        const __bf16* __restrict__ ktb = &kt[buf][0];
        const __bf16* __restrict__ vtb = &vt[buf][0];

        // ---- S = Q*(1/64) @ K^T : 4 tiles of 16x16, chained over d ----
        v8f s[4];
#pragma unroll
        for (int t = 0; t < 4; ++t) {
            v8f acc;
#pragma unroll
            for (int r = 0; r < 8; ++r) acc[r] = 0.0f;
#pragma unroll
            for (int c = 0; c < 2; ++c) {
                union { v16bf v; unsigned u[8]; } bt;
#pragma unroll
                for (int j = 0; j < 8; ++j) {
                    int d = c * 32 + (j & 3) * 2 + (j >> 2) * 16 + hi * 8;
                    bt.u[j] = *(const unsigned*)&ktb[(t * 16 + lo16) * KSTR + d];
                }
                acc = __builtin_amdgcn_wmma_f32_16x16x32_bf16(
                        false, aq[c], false, bt.v, (short)0, acc, false, false);
            }
            s[t] = acc;
        }

        // ---- mask invalid keys (tail block) ----
#pragma unroll
        for (int t = 0; t < 4; ++t) {
            if (key0 + t * 16 + lo16 >= NTOK) {
#pragma unroll
                for (int r = 0; r < 8; ++r) s[t][r] = -3.0e38f;
            }
        }

        // ---- online softmax (row reductions stay inside 16-lane halves) ----
#pragma unroll
        for (int r = 0; r < 8; ++r) {
            float v = fmaxf(fmaxf(s[0][r], s[1][r]), fmaxf(s[2][r], s[3][r]));
#pragma unroll
            for (int m = 1; m < 16; m <<= 1) v = fmaxf(v, __shfl_xor(v, m, 32));
            float mn = fmaxf(mrow[r], v);
            float f  = __expf(mrow[r] - mn);
            lrow[r] *= f;
#pragma unroll
            for (int t = 0; t < 4; ++t) oacc[t][r] *= f;
            mrow[r] = mn;
        }
#pragma unroll
        for (int t = 0; t < 4; ++t)
#pragma unroll
            for (int r = 0; r < 8; ++r) s[t][r] = __expf(s[t][r] - mrow[r]);
#pragma unroll
        for (int r = 0; r < 8; ++r) {
            float sum = (s[0][r] + s[1][r]) + (s[2][r] + s[3][r]);
#pragma unroll
            for (int m = 1; m < 16; m <<= 1) sum += __shfl_xor(sum, m, 32);
            lrow[r] += sum;
        }

        // ---- P -> wave-private LDS patch (C-layout -> A-layout via LDS) ----
        __bf16* pw = &pt[wave * 16 * PSTR];
#pragma unroll
        for (int t = 0; t < 4; ++t)
#pragma unroll
            for (int r = 0; r < 8; ++r)
                pw[(r + 8 * hi) * PSTR + t * 16 + lo16] = (__bf16)s[t][r];
        // in-wave ds store->load ordering handled by DScnt waits

        v16bf ap[2];
        {
            union { v16bf v; unsigned u[8]; } tmp;
#pragma unroll
            for (int c = 0; c < 2; ++c) {
#pragma unroll
                for (int j = 0; j < 8; ++j) {
                    int k = c * 32 + (j & 3) * 2 + (j >> 2) * 16 + hi * 8;
                    tmp.u[j] = *(const unsigned*)&pw[lo16 * PSTR + k];
                }
                ap[c] = tmp.v;
            }
        }

        // ---- O += P @ V : 4 d-tiles, chained over keys ----
#pragma unroll
        for (int t = 0; t < 4; ++t) {
#pragma unroll
            for (int c = 0; c < 2; ++c) {
                union { v16bf v; unsigned u[8]; } bt;
#pragma unroll
                for (int j = 0; j < 8; ++j) {
                    int k = c * 32 + (j & 3) * 2 + (j >> 2) * 16 + hi * 8;   // key pair (contiguous in vt)
                    bt.u[j] = *(const unsigned*)&vtb[(t * 16 + lo16) * VSTR + k];
                }
                oacc[t] = __builtin_amdgcn_wmma_f32_16x16x32_bf16(
                            false, ap[c], false, bt.v, (short)0, oacc[t], false, false);
            }
        }

        __syncthreads();   // next-buffer staging complete; current buffer free
    }

    // ---- epilogue: O / l, scatter out[b][(d*8+head)][row] ----
#pragma unroll
    for (int r = 0; r < 8; ++r) {
        int row = row0 + wave * 16 + r + 8 * hi;
        if (row < NTOK) {
            float inv = 1.0f / lrow[r];
#pragma unroll
            for (int t = 0; t < 4; ++t) {
                int d = t * 16 + lo16;
                ob[(size_t)(d * NHEADS + head) * NTOK + row] = oacc[t][r] * inv;
            }
        }
    }
}

extern "C" void kernel_launch(void* const* d_in, const int* in_sizes, int n_in,
                              void* d_out, int out_size, void* d_ws, size_t ws_size,
                              hipStream_t stream) {
    (void)in_sizes; (void)n_in; (void)d_ws; (void)ws_size; (void)out_size;
    const float* x = (const float*)d_in[0];
    float* out = (float*)d_out;
    dim3 grid(NRB * NHEADS * BATCH);   // 13 * 8 * 4 = 416 workgroups
    Attention_77163382440391_kernel<<<grid, 256, 0, stream>>>(x, out);
}
